// Block_3255585210535
// MI455X (gfx1250) — matrix operand order repaired and checked by reading, not compiled, not run on hardware
//
#include <hip/hip_runtime.h>
#include <math.h>

// ---------------- CDNA5 WMMA types ----------------
typedef __attribute__((ext_vector_type(16))) _Float16 v16h;
typedef __attribute__((ext_vector_type(8)))  _Float16 v8h;
typedef __attribute__((ext_vector_type(8)))  float    v8f;

// ---------------- problem constants ----------------
constexpr int Bb  = 8;
constexpr int Nn  = 1024;
constexpr int BN  = Bb * Nn;          // 8192 nodes
constexpr int CA  = 128;              // scalar channels
constexpr int CV  = 64;               // vector channels
constexpr int CD  = 32;               // tensor channels
constexpr int RK  = 32;               // tensor-product rank
constexpr int NR  = 8;                // radial basis size
constexpr int DEG = 16;               // edges per source node (contiguous)
constexpr int E_  = BN * DEG;         // 131072 edges
constexpr float PI_F = 3.14159265358979323846f;

// =====================================================================
// WMMA GEMM:  Y[M,N] = alpha * A[M,K] * W[N,K]^T  (+ Y if ACCUM)
// Block = 128 threads = 4 waves; each wave owns a 16(M) x 32(N) tile.
// The 32xK B tile is staged once per block into LDS (f16, zero-padded),
// then read with 16B ds loads. Requires M % 64 == 0 (all call sites).
//   MODE 0: phys = r      MODE 2: gather, phys = idx[r/C]*C + (r%C)
// =====================================================================
template<int K, int MODE, bool ACCUM>
__global__ __launch_bounds__(128)
void k_gemm(const float* __restrict__ X, const float* __restrict__ W,
            float* __restrict__ Y, int M, int Nout,
            float alpha, int C, const int* __restrict__ idx)
{
    constexpr int KITER = (K + 31) / 32;
    constexpr int LDSK  = KITER * 32;          // zero-padded K in LDS
    __shared__ _Float16 sB[32 * LDSK];

    const int wave = threadIdx.x >> 5;
    const int lane = threadIdx.x & 31;
    const int tm   = blockIdx.x * 64 + wave * 16;
    const int tn   = blockIdx.y * 32;
    const int hi   = lane >> 4;
    const int l15  = lane & 15;
    const int hi8  = hi * 8, hi16 = hi * 16;

    // ---- stage B tile (32 cols x LDSK) into LDS, f16, zero-padded ----
    {
        constexpr int PER = LDSK / 4;          // 4 threads per column
        const int col = threadIdx.x >> 2;
        const int kk0 = (threadIdx.x & 3) * PER;
        const float* wp = W + (size_t)(tn + col) * K;
#pragma unroll
        for (int q = 0; q < PER; ++q) {
            const int k = kk0 + q;
            if constexpr (K == LDSK) {
                sB[col * LDSK + k] = (_Float16)wp[k];
            } else {
                const int ks = (k < K) ? k : 0;
                float v = wp[ks];
                sB[col * LDSK + k] = (_Float16)((k < K) ? v : 0.f);
            }
        }
    }
    __syncthreads();

    // ---- map this lane's A row (M index = lane&15 per ISA A layout) ----
    const int r = tm + l15;
    long phys = r;
    if constexpr (MODE == 2) {
        int g = r / C, comp = r - g * C;
        phys = (long)idx[g] * C + comp;
    }
    const float* xrow = X + (size_t)phys * K;
    const _Float16* lb0 = sB + (size_t)l15 * LDSK;
    const _Float16* lb1 = sB + (size_t)(16 + l15) * LDSK;

    v8f acc0 = {}, acc1 = {};
#pragma unroll
    for (int it = 0; it < KITER; ++it) {
        const int k0 = it * 32;
        v16h a, b0, b1;
#pragma unroll
        for (int g = 0; g < 2; ++g) {
            const int gb = k0 + g * 16;        // compile-time group base
#pragma unroll
            for (int wq = 0; wq < 8; ++wq) {
                const int e = g * 8 + wq;
                float v;
                if (gb + 16 <= K)      v = xrow[gb + hi8 + wq];        // both halves
                else if (gb + 8 <= K)  v = hi ? 0.f : xrow[gb + wq];   // low half only
                else                   v = 0.f;
                a[e] = (_Float16)v;
            }
        }
        v8h q00 = *(const v8h*)(lb0 + k0 + hi8);
        v8h q01 = *(const v8h*)(lb0 + k0 + 16 + hi8);
        v8h q10 = *(const v8h*)(lb1 + k0 + hi8);
        v8h q11 = *(const v8h*)(lb1 + k0 + 16 + hi8);
#pragma unroll
        for (int e = 0; e < 8; ++e) {
            b0[e] = q00[e]; b0[e + 8] = q01[e];
            b1[e] = q10[e]; b1[e + 8] = q11[e];
        }
        acc0 = __builtin_amdgcn_wmma_f32_16x16x32_f16(false, a, false, b0,
                                                      (short)0, acc0, false, false);
        acc1 = __builtin_amdgcn_wmma_f32_16x16x32_f16(false, a, false, b1,
                                                      (short)0, acc1, false, false);
    }
    // ---- epilogue: gather previous Y first (clause), then fma, then store ----
    float p0[8], p1[8];
    if constexpr (ACCUM) {
#pragma unroll
        for (int e = 0; e < 8; ++e) {
            const size_t o = (size_t)(tm + e + hi8) * Nout + tn + l15;
            p0[e] = Y[o]; p1[e] = Y[o + 16];
        }
    }
#pragma unroll
    for (int e = 0; e < 8; ++e) {
        const size_t o = (size_t)(tm + e + hi8) * Nout + tn + l15;
        float r0 = alpha * acc0[e];
        float r1 = alpha * acc1[e];
        if constexpr (ACCUM) { r0 += p0[e]; r1 += p1[e]; }
        Y[o]      = r0;
        Y[o + 16] = r1;
    }
}

template<int K, int MODE>
static inline void launch_g(hipStream_t s, const float* X, const float* W, float* Y,
                            int M, int N, float alpha, int accum, int C, const int* idx)
{
    dim3 g(M / 64, N / 32), b(128);
    if (accum) k_gemm<K,MODE,true> <<<g,b,0,s>>>(X,W,Y,M,N,alpha,C,idx);
    else       k_gemm<K,MODE,false><<<g,b,0,s>>>(X,W,Y,M,N,alpha,C,idx);
}

static void gemm(hipStream_t s, const float* X, const float* W, float* Y,
                 int M, int K, int N, float alpha, int accum,
                 int mode = 0, int C = 1, const int* idx = nullptr)
{
    if (mode == 2) {
        switch (K) {
        case 32:  launch_g<32,2> (s,X,W,Y,M,N,alpha,accum,C,idx); break;
        case 64:  launch_g<64,2> (s,X,W,Y,M,N,alpha,accum,C,idx); break;
        default:  launch_g<128,2>(s,X,W,Y,M,N,alpha,accum,C,idx); break;
        }
    } else {
        switch (K) {
        case 8:   launch_g<8,0>  (s,X,W,Y,M,N,alpha,accum,C,idx); break;
        case 32:  launch_g<32,0> (s,X,W,Y,M,N,alpha,accum,C,idx); break;
        case 64:  launch_g<64,0> (s,X,W,Y,M,N,alpha,accum,C,idx); break;
        default:  launch_g<128,0>(s,X,W,Y,M,N,alpha,accum,C,idx); break;
        }
    }
}

// =====================================================================
// Fused 7-tap conv over the node axis:  Y[r,o] (+)= sum_t sum_i
//   W[o,i,t] * X[row shifted by (t-3) within the batch sample, i]
// All 7 taps' 32xK B tiles staged in LDS (<= 56 KB); accumulators stay
// in VGPRs across taps -> one epilogue instead of 7 RMW passes.
// K = D (square conv). M % 64 == 0. C = #components (rows per node).
// =====================================================================
template<int K, bool ACCUM>
__global__ __launch_bounds__(128)
void k_conv(const float* __restrict__ X, const float* __restrict__ W,
            float* __restrict__ Y, int M, int C)
{
    constexpr int KITER = K / 32;
    __shared__ _Float16 sB[7 * 32 * K];

    const int wave = threadIdx.x >> 5;
    const int lane = threadIdx.x & 31;
    const int tm   = blockIdx.x * 64 + wave * 16;
    const int tn   = blockIdx.y * 32;
    const int hi   = lane >> 4;
    const int l15  = lane & 15;
    const int hi8  = hi * 8, hi16 = hi * 16;

    // ---- stage all 7 taps: sB[t][col][k] = W[(tn+col)*K*7 + k*7 + t] ----
    {
        constexpr int PER = K / 4;
        const int col = threadIdx.x >> 2;
        const int kk0 = (threadIdx.x & 3) * PER;
        const float* wp = W + (size_t)(tn + col) * (K * 7);
#pragma unroll
        for (int t = 0; t < 7; ++t)
#pragma unroll
            for (int q = 0; q < PER; ++q) {
                const int k = kk0 + q;
                sB[((size_t)t * 32 + col) * K + k] = (_Float16)wp[(size_t)k * 7 + t];
            }
    }
    __syncthreads();

    const int r = tm + l15;
    const int node = r / C;
    const int n = node % Nn;                 // position within batch sample

    v8f acc0 = {}, acc1 = {};
#pragma unroll
    for (int t = 0; t < 7; ++t) {
        const int d = t - 3;
        const bool valid = (unsigned)(n + d) < (unsigned)Nn;
        long pr = (long)r + (long)d * C;     // shifted row
        if (!valid) pr = r;                  // clamp in-bounds; value masked to 0
        const float* xrow = X + (size_t)pr * K;
        const _Float16* lb0 = sB + ((size_t)t * 32 + l15) * K;
        const _Float16* lb1 = sB + ((size_t)t * 32 + 16 + l15) * K;
#pragma unroll
        for (int it = 0; it < KITER; ++it) {
            const int k0 = it * 32;
            v16h a, b0, b1;
#pragma unroll
            for (int g = 0; g < 2; ++g)
#pragma unroll
                for (int wq = 0; wq < 8; ++wq) {
                    float v = xrow[k0 + g * 16 + hi8 + wq];
                    a[g * 8 + wq] = (_Float16)(valid ? v : 0.f);
                }
            v8h q00 = *(const v8h*)(lb0 + k0 + hi8);
            v8h q01 = *(const v8h*)(lb0 + k0 + 16 + hi8);
            v8h q10 = *(const v8h*)(lb1 + k0 + hi8);
            v8h q11 = *(const v8h*)(lb1 + k0 + 16 + hi8);
#pragma unroll
            for (int e = 0; e < 8; ++e) {
                b0[e] = q00[e]; b0[e + 8] = q01[e];
                b1[e] = q10[e]; b1[e + 8] = q11[e];
            }
            acc0 = __builtin_amdgcn_wmma_f32_16x16x32_f16(false, a, false, b0,
                                                          (short)0, acc0, false, false);
            acc1 = __builtin_amdgcn_wmma_f32_16x16x32_f16(false, a, false, b1,
                                                          (short)0, acc1, false, false);
        }
    }
    // ---- single epilogue ----
    float p0[8], p1[8];
    if constexpr (ACCUM) {
#pragma unroll
        for (int e = 0; e < 8; ++e) {
            const size_t o = (size_t)(tm + e + hi8) * K + tn + l15;
            p0[e] = Y[o]; p1[e] = Y[o + 16];
        }
    }
#pragma unroll
    for (int e = 0; e < 8; ++e) {
        const size_t o = (size_t)(tm + e + hi8) * K + tn + l15;
        float r0 = acc0[e];
        float r1 = acc1[e];
        if constexpr (ACCUM) { r0 += p0[e]; r1 += p1[e]; }
        Y[o]      = r0;
        Y[o + 16] = r1;
    }
}

static void conv7(hipStream_t s, const float* X, const float* W, float* Y,
                  int Cc, int D, int accum)
{
    const int M = BN * Cc;
    dim3 g(M / 64, D / 32), b(128);
    if (accum) {
        switch (D) {
        case 32:  k_conv<32,true>  <<<g,b,0,s>>>(X,W,Y,M,Cc); break;
        case 64:  k_conv<64,true>  <<<g,b,0,s>>>(X,W,Y,M,Cc); break;
        default:  k_conv<128,true> <<<g,b,0,s>>>(X,W,Y,M,Cc); break;
        }
    } else {
        switch (D) {
        case 32:  k_conv<32,false> <<<g,b,0,s>>>(X,W,Y,M,Cc); break;
        case 64:  k_conv<64,false> <<<g,b,0,s>>>(X,W,Y,M,Cc); break;
        default:  k_conv<128,false><<<g,b,0,s>>>(X,W,Y,M,Cc); break;
        }
    }
}

// =====================================================================
// Elementwise / small kernels
// =====================================================================
__global__ void k_prep_a(const float* __restrict__ xa, const float* __restrict__ p0,
                         const float* __restrict__ p1, const float* __restrict__ pos_a,
                         float* __restrict__ XA)
{
    int t = blockIdx.x * blockDim.x + threadIdx.x;
    if (t >= BN * CA) return;
    int node = t >> 7, c = t & 127;
    int n = node % Nn;
    float sarc = (float)n / (float)(Nn - 1);
    float d0 = 0.1f * (p1[node*3+0] - p0[node*3+0]);
    float d1 = 0.1f * (p1[node*3+1] - p0[node*3+1]);
    float d2 = 0.1f * (p1[node*3+2] - p0[node*3+2]);
    float dsq = d0*d0 + d1*d1 + d2*d2;
    XA[t] = xa[t] + cosf(sarc * PI_F * (float)(c + 1)) + pos_a[c] * dsq;
}

__global__ void k_prep_v(const float* __restrict__ xv, const float* __restrict__ p0,
                         const float* __restrict__ p1, const float* __restrict__ pos_v,
                         const float* __restrict__ nd_l, const float* __restrict__ nd_r,
                         float* __restrict__ XV)
{
    int t = blockIdx.x * blockDim.x + threadIdx.x;
    if (t >= BN * 3 * CV) return;
    int node = t / 192, rem = t % 192, comp = rem / CV, c = rem % CV;
    float v = xv[node*192 + c*3 + comp];
    float dp = 0.1f * (p1[node*3+comp] - p0[node*3+comp]);
    v += pos_v[c] * dp;
    int n = node % Nn;
    float dl0=0.f, dl1=0.f, dr0=0.f, dr1=0.f;
    if (n < Nn - 1) {
        dl0 = p0[(node+1)*3+comp] - p0[node*3+comp];
        dl1 = p1[(node+1)*3+comp] - p1[node*3+comp];
    }
    if (n > 0) {
        dr0 = p0[node*3+comp] - p0[(node-1)*3+comp];
        dr1 = p1[node*3+comp] - p1[(node-1)*3+comp];
    }
    v += 0.1f * (nd_l[c*2+0]*dl0 + nd_l[c*2+1]*dl1 +
                 nd_r[c*2+0]*dr0 + nd_r[c*2+1]*dr1);
    XV[t] = v;
}

__global__ void k_prep_d(const float* __restrict__ xd, float* __restrict__ XD)
{
    int t = blockIdx.x * blockDim.x + threadIdx.x;
    if (t >= BN * 9 * CD) return;
    int node = t / 288, rem = t % 288, ij = rem / CD, c = rem % CD;
    XD[t] = xd[node*288 + c*9 + ij];
}

__global__ void k_trace(const float* __restrict__ XD, float* __restrict__ TRD)
{
    int t = blockIdx.x * blockDim.x + threadIdx.x;
    if (t >= BN * CD) return;
    int node = t >> 5, c = t & 31;
    const float* b = XD + (size_t)node * 288;
    TRD[t] = b[0*CD + c] + b[4*CD + c] + b[8*CD + c];
}

__global__ void k_addiso(float* __restrict__ YD, const float* __restrict__ TA)
{
    int t = blockIdx.x * blockDim.x + threadIdx.x;
    if (t >= BN * CD) return;
    int node = t >> 5, c = t & 31;
    float v = TA[t] * (1.0f / 3.0f);
    float* b = YD + (size_t)node * 288 + c;
    b[0*CD] += v; b[4*CD] += v; b[8*CD] += v;
}

// out[node,m,c] = Y[node,m,c] + gamma[c]*groupnorm(D[node,:,:])
__global__ void k_gn_add(const float* __restrict__ Yb, const float* __restrict__ Db,
                         const float* __restrict__ gamma, float* __restrict__ out,
                         int Cc, int D, int order)
{
    int t = blockIdx.x * blockDim.x + threadIdx.x;
    if (t >= BN * 8) return;
    int node = t >> 3, g = t & 7;
    int chG = D >> 3;
    int nvals = chG * Cc;
    const float* base = Db + (size_t)node * Cc * D;
    float mean = 0.f;
    if (order == 0) {
        for (int m = 0; m < Cc; ++m)
            for (int cc = 0; cc < chG; ++cc) mean += base[m*D + g*chG + cc];
        mean /= (float)nvals;
    }
    float ss = 0.f;
    for (int m = 0; m < Cc; ++m)
        for (int cc = 0; cc < chG; ++cc) {
            float v = base[m*D + g*chG + cc] - mean;
            ss += v * v;
        }
    float inv = rsqrtf(ss / (float)nvals + 1e-5f);
    for (int m = 0; m < Cc; ++m)
        for (int cc = 0; cc < chG; ++cc) {
            int c = g*chG + cc;
            size_t o = (size_t)node * Cc * D + m*D + c;
            out[o] = Yb[o] + (base[m*D + c] - mean) * inv * gamma[c];
        }
}

__global__ void k_leaky(float* __restrict__ buf, int total)
{
    int t = blockIdx.x * blockDim.x + threadIdx.x;
    if (t >= total) return;
    float v = buf[t];
    buf[t] = (v >= 0.f) ? v : 0.1f * v;
}

// gated norm sigmoid over components:  x *= tanh(|x|/2)/|x|
__global__ void k_vsig(float* __restrict__ buf, int Cc, int D)
{
    int t = blockIdx.x * blockDim.x + threadIdx.x;
    if (t >= BN * D) return;
    int c = t % D, node = t / D;
    float* b = buf + (size_t)node * Cc * D + c;
    float n2 = 1e-12f;
    for (int m = 0; m < Cc; ++m) { float v = b[m*D]; n2 += v * v; }
    float nn = sqrtf(n2);
    float sc = tanhf(0.5f * nn) / nn;
    for (int m = 0; m < Cc; ++m) b[m*D] *= sc;
}

// edge geometry + radial basis + gated direction
__global__ void k_edge(const float* __restrict__ pos, const float* __restrict__ box,
                       const int* __restrict__ src, const int* __restrict__ dst,
                       float* __restrict__ rad, float* __restrict__ rs)
{
    int e = blockIdx.x * blockDim.x + threadIdx.x;
    if (e >= E_) return;
    int i = src[e], j = dst[e];
    float bx = box[0], by = box[1], bz = box[2];
    float rx = pos[j*3+0] - pos[i*3+0]; rx -= bx * rintf(rx / bx);
    float ry = pos[j*3+1] - pos[i*3+1]; ry -= by * rintf(ry / by);
    float rz = pos[j*3+2] - pos[i*3+2]; rz -= bz * rintf(rz / bz);
    float xsq = (rx*rx + ry*ry + rz*rz) * 0.5f;            // |r|^2 / R0, R0=2
    float env = fmaxf(1.f - xsq, 0.f);
    float arg = sqrtf(xsq + 1e-12f);
    for (int k = 0; k < NR; ++k)
        rad[(size_t)e*NR + k] = cosf(PI_F * (float)k * arg) * env;
    float sx = rx * 8.5f, sy = ry * 8.5f, sz = rz * 8.5f;  // 17/R0
    float nn = sqrtf(sx*sx + sy*sy + sz*sz + 1e-12f);
    float sc = tanhf(0.5f * nn) / nn;
    rs[(size_t)e*3+0] = sx * sc;
    rs[(size_t)e*3+1] = sy * sc;
    rs[(size_t)e*3+2] = sz * sc;
}

// per-node ACE moment sums over the 16 contiguous edges
__global__ void k_ace(const float* __restrict__ rad, const float* __restrict__ rs,
                      float* __restrict__ Aa, float* __restrict__ Av, float* __restrict__ Ad)
{
    int t = blockIdx.x * blockDim.x + threadIdx.x;
    if (t >= BN * NR) return;
    int node = t >> 3, k = t & 7;
    float sa = 0.f, sv[3] = {0,0,0}, sd[9] = {0,0,0,0,0,0,0,0,0};
    for (int q = 0; q < DEG; ++q) {
        int e = node * DEG + q;
        float rv = rad[(size_t)e*NR + k];
        float s0 = rs[(size_t)e*3+0], s1 = rs[(size_t)e*3+1], s2 = rs[(size_t)e*3+2];
        sa += rv;
        sv[0] += rv*s0; sv[1] += rv*s1; sv[2] += rv*s2;
        sd[0] += rv*s0*s0; sd[1] += rv*s0*s1; sd[2] += rv*s0*s2;
        sd[3] += rv*s1*s0; sd[4] += rv*s1*s1; sd[5] += rv*s1*s2;
        sd[6] += rv*s2*s0; sd[7] += rv*s2*s1; sd[8] += rv*s2*s2;
    }
    Aa[(size_t)node*NR + k] = sa;
    for (int m = 0; m < 3; ++m) Av[((size_t)node*3 + m)*NR + k] = sv[m];
    for (int ij = 0; ij < 9; ++ij) Ad[((size_t)node*9 + ij)*NR + k] = sd[ij];
}

// yrv[(e,m),c] = wr[e,c] * rs[e,m]
__global__ void k_expand(const float* __restrict__ wr, const float* __restrict__ rsc,
                         float* __restrict__ yrv, int Ec)
{
    int t = blockIdx.x * blockDim.x + threadIdx.x;
    if (t >= Ec * 96) return;
    int c = t & 31, rest = t >> 5, m = rest % 3, e = rest / 3;
    yrv[t] = wr[(size_t)e*32 + c] * rsc[(size_t)e*3 + m];
}

// Clebsch-Gordan products (rank channel = 32)
__global__ void k_cg(const float* __restrict__ XL, const float* __restrict__ YR,
                     float* __restrict__ P, int n, int op, int Cc)
{
    int t = blockIdx.x * blockDim.x + threadIdx.x;
    int c = t & 31, rest = t >> 5;
    switch (op) {
    case 0: { // elementwise; n = total rows
        if (rest >= n) return;
        P[t] = XL[t] * YR[t];
    } break;
    case 1: { // dot over Cc components
        if (rest >= n) return;
        float s = 0.f;
        for (int m = 0; m < Cc; ++m)
            s += XL[((size_t)rest*Cc + m)*32 + c] * YR[((size_t)rest*Cc + m)*32 + c];
        P[t] = s;
    } break;
    case 2: { // scalarL x vecR
        int g = rest / Cc; if (g >= n) return;
        P[t] = XL[(size_t)g*32 + c] * YR[t];
    } break;
    case 3: { // vecL x scalarR
        int g = rest / Cc; if (g >= n) return;
        P[t] = XL[t] * YR[(size_t)g*32 + c];
    } break;
    case 4: { // matvec (2,1->1)
        int i = rest % 3, g = rest / 3; if (g >= n) return;
        float s = 0.f;
        for (int j = 0; j < 3; ++j)
            s += XL[((size_t)g*9 + i*3 + j)*32 + c] * YR[((size_t)g*3 + j)*32 + c];
        P[t] = s;
    } break;
    case 5: { // outer (1,1->2)
        int ij = rest % 9, g = rest / 9; if (g >= n) return;
        int i = ij / 3, j = ij % 3;
        P[t] = XL[((size_t)g*3 + i)*32 + c] * YR[((size_t)g*3 + j)*32 + c];
    } break;
    case 6: { // matmat (2,2->2)
        int ij = rest % 9, g = rest / 9; if (g >= n) return;
        int i = ij / 3, j = ij % 3;
        float s = 0.f;
        for (int k = 0; k < 3; ++k)
            s += XL[((size_t)g*9 + i*3 + k)*32 + c] * YR[((size_t)g*9 + k*3 + j)*32 + c];
        P[t] = s;
    } break;
    }
}

// contiguous 16-edge segment sum
__global__ void k_segsum(const float* __restrict__ P, float* __restrict__ S,
                         int nodes, int Cc)
{
    int t = blockIdx.x * blockDim.x + threadIdx.x;
    if (t >= nodes * Cc * 32) return;
    int c = t & 31, rest = t >> 5, m = rest % Cc, node = rest / Cc;
    float s = 0.f;
    for (int q = 0; q < DEG; ++q)
        s += P[(((size_t)(node*DEG + q))*Cc + m)*32 + c];
    S[t] = s;
}

__global__ void k_head(const float* __restrict__ T, const float* __restrict__ E2,
                       float* __restrict__ out)
{
    int node = blockIdx.x * blockDim.x + threadIdx.x;
    if (node >= BN) return;
    float s = 0.f;
    for (int c = 0; c < CA; ++c) s += E2[c] * T[(size_t)node*CA + c];
    out[node] = s;
}

// =====================================================================
// Host-side helpers
// =====================================================================
static inline dim3 g1(int total) { return dim3((total + 255) / 256); }

static void cg(hipStream_t s, const float* XL, const float* YR, float* P,
               int n, int op, int Cc)
{
    int total;
    switch (op) {
    case 0: case 1: total = n * 32; break;
    case 2: case 3: total = n * Cc * 32; break;
    case 4: total = n * 3 * 32; break;
    default: total = n * 9 * 32; break;
    }
    k_cg<<<g1(total), 256, 0, s>>>(XL, YR, P, n, op, Cc);
}

static void segsum(hipStream_t s, const float* P, float* S, int nodes, int Cc)
{
    k_segsum<<<g1(nodes * Cc * 32), 256, 0, s>>>(P, S, nodes, Cc);
}

struct TP   { const float *WL, *WR, *WO; };
struct SubP {
    const float *Laa, *Lad, *Lvv, *Ldd, *Lda;
    TP aaa, vva, dda, avv, dvv, vvd, add_, ddd;
    const float *ga, *gv, *gd;
    const float *conv_a[3], *conv_v[3], *conv_d[3];
    const float *mlp_a[3],  *mlp_v[3],  *mlp_d[3];
};
struct MesP { TP t000, t101, t202, t011, t110, t112, t211; };

static const float* F(void* const* d, int& i) { return (const float*)d[i++]; }
static TP getTP(void* const* d, int& i) {
    TP t; t.WL = F(d,i); t.WR = F(d,i); t.WO = F(d,i); return t;
}
static SubP getSub(void* const* d, int& i) {
    SubP p;
    p.Laa=F(d,i); p.Lad=F(d,i); p.Lvv=F(d,i); p.Ldd=F(d,i); p.Lda=F(d,i);
    p.aaa=getTP(d,i); p.vva=getTP(d,i); p.dda=getTP(d,i); p.avv=getTP(d,i);
    p.dvv=getTP(d,i); p.vvd=getTP(d,i); p.add_=getTP(d,i); p.ddd=getTP(d,i);
    p.ga=F(d,i); p.gv=F(d,i); p.gd=F(d,i);
    for (int k=0;k<3;++k) p.conv_a[k]=F(d,i);
    for (int k=0;k<3;++k) p.conv_v[k]=F(d,i);
    for (int k=0;k<3;++k) p.conv_d[k]=F(d,i);
    for (int k=0;k<3;++k) p.mlp_a[k]=F(d,i);
    for (int k=0;k<3;++k) p.mlp_v[k]=F(d,i);
    for (int k=0;k<3;++k) p.mlp_d[k]=F(d,i);
    return p;
}
static MesP getMes(void* const* d, int& i) {
    MesP m;
    m.t000=getTP(d,i); m.t101=getTP(d,i); m.t202=getTP(d,i); m.t011=getTP(d,i);
    m.t110=getTP(d,i); m.t112=getTP(d,i); m.t211=getTP(d,i);
    return m;
}

struct WS {
    float *XA,*XV,*XD, *YA,*YV,*YD, *DAb,*DVb,*DDb, *XNA,*XNV,*XND;
    float *T1,*T2, *XL,*YR,*P, *TRD,*TA;
    float *rad0,*rs0,*rad1,*rs1, *Aa,*Av,*Ad;
    float *XLE,*WRr,*YRV,*PE,*SSb;
};

static void subblock(hipStream_t s, const SubP& p, WS& w)
{
    const int SVr = BN*3, SDr = BN*9;
    // linear mix (ya/yv/yd), residual stays in XA/XV/XD
    k_trace<<<g1(BN*CD), 256, 0, s>>>(w.XD, w.TRD);
    gemm(s, w.XA,  p.Laa, w.YA, BN, CA, CA, 1.f, 0);
    gemm(s, w.TRD, p.Lad, w.YA, BN, CD, CA, 1.f, 1);
    gemm(s, w.XV,  p.Lvv, w.YV, SVr, CV, CV, 1.f, 0);
    gemm(s, w.XD,  p.Ldd, w.YD, SDr, CD, CD, 1.f, 0);
    gemm(s, w.XA,  p.Lda, w.TA, BN, CA, CD, 1.f, 0);
    k_addiso<<<g1(BN*CD), 256, 0, s>>>(w.YD, w.TA);

    // tensor products on 0.3*y (alpha folded into WL/WR gemms)
    // -> da (scalar)
    gemm(s, w.YA, p.aaa.WL, w.XL, BN, CA, RK, 0.3f, 0);
    gemm(s, w.YA, p.aaa.WR, w.YR, BN, CA, RK, 0.3f, 0);
    cg(s, w.XL, w.YR, w.P, BN, 0, 1);
    gemm(s, w.P, p.aaa.WO, w.DAb, BN, RK, CA, 1.f, 0);
    gemm(s, w.YV, p.vva.WL, w.XL, SVr, CV, RK, 0.3f, 0);
    gemm(s, w.YV, p.vva.WR, w.YR, SVr, CV, RK, 0.3f, 0);
    cg(s, w.XL, w.YR, w.P, BN, 1, 3);
    gemm(s, w.P, p.vva.WO, w.DAb, BN, RK, CA, 1.f, 1);
    gemm(s, w.YD, p.dda.WL, w.XL, SDr, CD, RK, 0.3f, 0);
    gemm(s, w.YD, p.dda.WR, w.YR, SDr, CD, RK, 0.3f, 0);
    cg(s, w.XL, w.YR, w.P, BN, 1, 9);
    gemm(s, w.P, p.dda.WO, w.DAb, BN, RK, CA, 1.f, 1);
    // -> dv (vector)
    gemm(s, w.YA, p.avv.WL, w.XL, BN, CA, RK, 0.3f, 0);
    gemm(s, w.YV, p.avv.WR, w.YR, SVr, CV, RK, 0.3f, 0);
    cg(s, w.XL, w.YR, w.P, BN, 2, 3);
    gemm(s, w.P, p.avv.WO, w.DVb, SVr, RK, CV, 1.f, 0);
    gemm(s, w.YD, p.dvv.WL, w.XL, SDr, CD, RK, 0.3f, 0);
    gemm(s, w.YV, p.dvv.WR, w.YR, SVr, CV, RK, 0.3f, 0);
    cg(s, w.XL, w.YR, w.P, BN, 4, 3);
    gemm(s, w.P, p.dvv.WO, w.DVb, SVr, RK, CV, 1.f, 1);
    // -> dd (tensor)
    gemm(s, w.YV, p.vvd.WL, w.XL, SVr, CV, RK, 0.3f, 0);
    gemm(s, w.YV, p.vvd.WR, w.YR, SVr, CV, RK, 0.3f, 0);
    cg(s, w.XL, w.YR, w.P, BN, 5, 3);
    gemm(s, w.P, p.vvd.WO, w.DDb, SDr, RK, CD, 1.f, 0);
    gemm(s, w.YA, p.add_.WL, w.XL, BN, CA, RK, 0.3f, 0);
    gemm(s, w.YD, p.add_.WR, w.YR, SDr, CD, RK, 0.3f, 0);
    cg(s, w.XL, w.YR, w.P, BN, 2, 9);
    gemm(s, w.P, p.add_.WO, w.DDb, SDr, RK, CD, 1.f, 1);
    gemm(s, w.YD, p.ddd.WL, w.XL, SDr, CD, RK, 0.3f, 0);
    gemm(s, w.YD, p.ddd.WR, w.YR, SDr, CD, RK, 0.3f, 0);
    cg(s, w.XL, w.YR, w.P, BN, 6, 9);
    gemm(s, w.P, p.ddd.WO, w.DDb, SDr, RK, CD, 1.f, 1);

    // x_new = y + groupnorm(d)
    k_gn_add<<<g1(BN*8), 256, 0, s>>>(w.YA, w.DAb, p.ga, w.XNA, 1, CA, 0);
    k_gn_add<<<g1(BN*8), 256, 0, s>>>(w.YV, w.DVb, p.gv, w.XNV, 3, CV, 1);
    k_gn_add<<<g1(BN*8), 256, 0, s>>>(w.YD, w.DDb, p.gd, w.XND, 9, CD, 2);

    // out = residual(X) + mlp(x_new) + conv(x_new); accumulate into X
    // ---- scalar ----
    gemm(s, w.XNA, p.mlp_a[0], w.T1, BN, CA, CA, 1.f, 0);
    k_leaky<<<g1(BN*CA), 256, 0, s>>>(w.T1, BN*CA);
    gemm(s, w.T1, p.mlp_a[1], w.T2, BN, CA, CA, 1.f, 0);
    k_leaky<<<g1(BN*CA), 256, 0, s>>>(w.T2, BN*CA);
    gemm(s, w.T2, p.mlp_a[2], w.XA, BN, CA, CA, 1.f, 1);
    conv7(s, w.XNA, p.conv_a[0], w.T1, 1, CA, 0);
    k_leaky<<<g1(BN*CA), 256, 0, s>>>(w.T1, BN*CA);
    conv7(s, w.T1, p.conv_a[1], w.T2, 1, CA, 0);
    k_leaky<<<g1(BN*CA), 256, 0, s>>>(w.T2, BN*CA);
    conv7(s, w.T2, p.conv_a[2], w.XA, 1, CA, 1);
    // ---- vector ----
    gemm(s, w.XNV, p.mlp_v[0], w.T1, SVr, CV, CV, 1.f, 0);
    k_vsig<<<g1(BN*CV), 256, 0, s>>>(w.T1, 3, CV);
    gemm(s, w.T1, p.mlp_v[1], w.T2, SVr, CV, CV, 1.f, 0);
    k_vsig<<<g1(BN*CV), 256, 0, s>>>(w.T2, 3, CV);
    gemm(s, w.T2, p.mlp_v[2], w.XV, SVr, CV, CV, 1.f, 1);
    conv7(s, w.XNV, p.conv_v[0], w.T1, 3, CV, 0);
    k_vsig<<<g1(BN*CV), 256, 0, s>>>(w.T1, 3, CV);
    conv7(s, w.T1, p.conv_v[1], w.T2, 3, CV, 0);
    k_vsig<<<g1(BN*CV), 256, 0, s>>>(w.T2, 3, CV);
    conv7(s, w.T2, p.conv_v[2], w.XV, 3, CV, 1);
    // ---- tensor ----
    gemm(s, w.XND, p.mlp_d[0], w.T1, SDr, CD, CD, 1.f, 0);
    k_vsig<<<g1(BN*CD), 256, 0, s>>>(w.T1, 9, CD);
    gemm(s, w.T1, p.mlp_d[1], w.T2, SDr, CD, CD, 1.f, 0);
    k_vsig<<<g1(BN*CD), 256, 0, s>>>(w.T2, 9, CD);
    gemm(s, w.T2, p.mlp_d[2], w.XD, SDr, CD, CD, 1.f, 1);
    conv7(s, w.XND, p.conv_d[0], w.T1, 9, CD, 0);
    k_vsig<<<g1(BN*CD), 256, 0, s>>>(w.T1, 9, CD);
    conv7(s, w.T1, p.conv_d[1], w.T2, 9, CD, 0);
    k_vsig<<<g1(BN*CD), 256, 0, s>>>(w.T2, 9, CD);
    conv7(s, w.T2, p.conv_d[2], w.XD, 9, CD, 1);
}

// Message pass: WO commuted past the (contiguous) segment sum.
static void message(hipStream_t s, const MesP& m, const float* rad, const float* rs,
                    const int* edst, WS& w)
{
    const float *SXA = w.YA, *SXV = w.YV, *SXD = w.YD;  // snapshot (caller fills)
    const int NC = 1024, Ec = NC * DEG;
    for (int n0 = 0; n0 < BN; n0 += NC) {
        int e0 = n0 * DEG;
        const int*   idx  = edst + e0;
        const float* radc = rad + (size_t)e0 * NR;
        const float* rsc  = rs  + (size_t)e0 * 3;
        float* XAo = w.XA + (size_t)n0 * CA;
        float* XVo = w.XV + (size_t)n0 * 3 * CV;
        float* XDo = w.XD + (size_t)n0 * 9 * CD;
        // tp_000 : (a x rad -> a)
        gemm(s, SXA, m.t000.WL, w.XLE, Ec, CA, RK, 1.f, 0, 2, 1, idx);
        gemm(s, radc, m.t000.WR, w.WRr, Ec, NR, RK, 1.f, 0);
        cg(s, w.XLE, w.WRr, w.PE, Ec, 0, 1);
        segsum(s, w.PE, w.SSb, NC, 1);
        gemm(s, w.SSb, m.t000.WO, XAo, NC, RK, CA, 1.f, 1);
        // tp_110 : (v x phi_v -> a)
        gemm(s, SXV, m.t110.WL, w.XLE, Ec*3, CV, RK, 1.f, 0, 2, 3, idx);
        gemm(s, radc, m.t110.WR, w.WRr, Ec, NR, RK, 1.f, 0);
        k_expand<<<g1(Ec*96), 256, 0, s>>>(w.WRr, rsc, w.YRV, Ec);
        cg(s, w.XLE, w.YRV, w.PE, Ec, 1, 3);
        segsum(s, w.PE, w.SSb, NC, 1);
        gemm(s, w.SSb, m.t110.WO, XAo, NC, RK, CA, 1.f, 1);
        // tp_011 : (a x phi_v -> v)
        gemm(s, SXA, m.t011.WL, w.XLE, Ec, CA, RK, 1.f, 0, 2, 1, idx);
        gemm(s, radc, m.t011.WR, w.WRr, Ec, NR, RK, 1.f, 0);
        k_expand<<<g1(Ec*96), 256, 0, s>>>(w.WRr, rsc, w.YRV, Ec);
        cg(s, w.XLE, w.YRV, w.PE, Ec, 2, 3);
        segsum(s, w.PE, w.SSb, NC, 3);
        gemm(s, w.SSb, m.t011.WO, XVo, NC*3, RK, CV, 1.f, 1);
        // tp_101 : (v x rad -> v)
        gemm(s, SXV, m.t101.WL, w.XLE, Ec*3, CV, RK, 1.f, 0, 2, 3, idx);
        gemm(s, radc, m.t101.WR, w.WRr, Ec, NR, RK, 1.f, 0);
        cg(s, w.XLE, w.WRr, w.PE, Ec, 3, 3);
        segsum(s, w.PE, w.SSb, NC, 3);
        gemm(s, w.SSb, m.t101.WO, XVo, NC*3, RK, CV, 1.f, 1);
        // tp_211 : (d x phi_v -> v)
        gemm(s, SXD, m.t211.WL, w.XLE, Ec*9, CD, RK, 1.f, 0, 2, 9, idx);
        gemm(s, radc, m.t211.WR, w.WRr, Ec, NR, RK, 1.f, 0);
        k_expand<<<g1(Ec*96), 256, 0, s>>>(w.WRr, rsc, w.YRV, Ec);
        cg(s, w.XLE, w.YRV, w.PE, Ec, 4, 3);
        segsum(s, w.PE, w.SSb, NC, 3);
        gemm(s, w.SSb, m.t211.WO, XVo, NC*3, RK, CV, 1.f, 1);
        // tp_112 : (v x phi_v -> d)
        gemm(s, SXV, m.t112.WL, w.XLE, Ec*3, CV, RK, 1.f, 0, 2, 3, idx);
        gemm(s, radc, m.t112.WR, w.WRr, Ec, NR, RK, 1.f, 0);
        k_expand<<<g1(Ec*96), 256, 0, s>>>(w.WRr, rsc, w.YRV, Ec);
        cg(s, w.XLE, w.YRV, w.PE, Ec, 5, 3);
        segsum(s, w.PE, w.SSb, NC, 9);
        gemm(s, w.SSb, m.t112.WO, XDo, NC*9, RK, CD, 1.f, 1);
        // tp_202 : (d x rad -> d)
        gemm(s, SXD, m.t202.WL, w.XLE, Ec*9, CD, RK, 1.f, 0, 2, 9, idx);
        gemm(s, radc, m.t202.WR, w.WRr, Ec, NR, RK, 1.f, 0);
        cg(s, w.XLE, w.WRr, w.PE, Ec, 3, 9);
        segsum(s, w.PE, w.SSb, NC, 9);
        gemm(s, w.SSb, m.t202.WO, XDo, NC*9, RK, CD, 1.f, 1);
    }
}

// =====================================================================
// kernel_launch
// =====================================================================
extern "C" void kernel_launch(void* const* d_in, const int* in_sizes, int n_in,
                              void* d_out, int out_size, void* d_ws, size_t ws_size,
                              hipStream_t stream)
{
    (void)in_sizes; (void)n_in; (void)out_size; (void)ws_size;
    int i = 0;
    const float* pos0  = F(d_in, i);
    const float* pos1  = F(d_in, i);
    const float* xa_in = F(d_in, i);
    const float* xv_in = F(d_in, i);
    const float* xd_in = F(d_in, i);
    const float* box   = F(d_in, i);
    const int* esrc = (const int*)d_in[i++];
    const int* edst = (const int*)d_in[i++];
    const float* pos_a = F(d_in, i);
    const float* pos_v = F(d_in, i);
    const float* nd_l  = F(d_in, i);
    const float* nd_r  = F(d_in, i);
    const float* ace0a = F(d_in, i); const float* ace0v = F(d_in, i); const float* ace0d = F(d_in, i);
    const float* ace1a = F(d_in, i); const float* ace1v = F(d_in, i); const float* ace1d = F(d_in, i);
    SubP sub0 = getSub(d_in, i);
    SubP sub1 = getSub(d_in, i);
    SubP sub2 = getSub(d_in, i);
    MesP mes0 = getMes(d_in, i);
    MesP mes1 = getMes(d_in, i);
    const float* lin_out = F(d_in, i);
    const float* E1p     = F(d_in, i);
    const float* E2p     = F(d_in, i);

    // ---- workspace bump allocator (needs ~200 MB) ----
    char* base = (char*)d_ws;
    size_t off = 0;
    auto alloc = [&](size_t nf) -> float* {
        float* p = (float*)(base + off);
        off += ((nf * sizeof(float) + 255) / 256) * 256;
        return p;
    };
    const size_t SA = (size_t)BN*CA, SV = (size_t)BN*3*CV, SD = (size_t)BN*9*CD;
    const int NC = 1024, Ec = NC * DEG;
    WS w;
    w.XA = alloc(SA); w.XV = alloc(SV); w.XD = alloc(SD);
    w.YA = alloc(SA); w.YV = alloc(SV); w.YD = alloc(SD);
    w.DAb = alloc(SA); w.DVb = alloc(SV); w.DDb = alloc(SD);
    w.XNA = alloc(SA); w.XNV = alloc(SV); w.XND = alloc(SD);
    w.T1 = alloc(SD); w.T2 = alloc(SD);
    w.XL = alloc(SD); w.YR = alloc(SD); w.P = alloc(SD);
    w.TRD = alloc((size_t)BN*CD); w.TA = alloc((size_t)BN*CD);
    w.rad0 = alloc((size_t)E_*NR); w.rs0 = alloc((size_t)E_*3);
    w.rad1 = alloc((size_t)E_*NR); w.rs1 = alloc((size_t)E_*3);
    w.Aa = alloc((size_t)BN*NR); w.Av = alloc((size_t)BN*3*NR); w.Ad = alloc((size_t)BN*9*NR);
    w.XLE = alloc((size_t)Ec*9*RK); w.WRr = alloc((size_t)Ec*RK);
    w.YRV = alloc((size_t)Ec*3*RK); w.PE = alloc((size_t)Ec*9*RK);
    w.SSb = alloc((size_t)NC*9*RK);

    // ---- embeds ----
    k_prep_a<<<g1(BN*CA), 256, 0, stream>>>(xa_in, pos0, pos1, pos_a, w.XA);
    k_prep_v<<<g1(BN*3*CV), 256, 0, stream>>>(xv_in, pos0, pos1, pos_v, nd_l, nd_r, w.XV);
    k_prep_d<<<g1(BN*9*CD), 256, 0, stream>>>(xd_in, w.XD);

    // ---- SubBlock 0 ----
    subblock(stream, sub0, w);

    // ---- edge geometry (both frames) ----
    k_edge<<<g1(E_), 256, 0, stream>>>(pos0, box, esrc, edst, w.rad0, w.rs0);
    k_edge<<<g1(E_), 256, 0, stream>>>(pos1, box, esrc, edst, w.rad1, w.rs1);

    // ---- ACE embeds ----
    for (int f = 0; f < 2; ++f) {
        const float* rad = f ? w.rad1 : w.rad0;
        const float* rs  = f ? w.rs1  : w.rs0;
        const float* pa  = f ? ace1a : ace0a;
        const float* pv  = f ? ace1v : ace0v;
        const float* pd  = f ? ace1d : ace0d;
        k_ace<<<g1(BN*NR), 256, 0, stream>>>(rad, rs, w.Aa, w.Av, w.Ad);
        gemm(stream, w.Aa, pa, w.XA, BN,   NR, CA, 1.f, 1);
        gemm(stream, w.Av, pv, w.XV, BN*3, NR, CV, 1.f, 1);
        gemm(stream, w.Ad, pd, w.XD, BN*9, NR, CD, 1.f, 1);
    }

    // ---- SubBlock 1 ----
    subblock(stream, sub1, w);

    // ---- ACE message passes (snapshot x so gathers see pre-update state) ----
    hipMemcpyAsync(w.YA, w.XA, SA*4, hipMemcpyDeviceToDevice, stream);
    hipMemcpyAsync(w.YV, w.XV, SV*4, hipMemcpyDeviceToDevice, stream);
    hipMemcpyAsync(w.YD, w.XD, SD*4, hipMemcpyDeviceToDevice, stream);
    message(stream, mes0, w.rad0, w.rs0, edst, w);
    hipMemcpyAsync(w.YA, w.XA, SA*4, hipMemcpyDeviceToDevice, stream);
    hipMemcpyAsync(w.YV, w.XV, SV*4, hipMemcpyDeviceToDevice, stream);
    hipMemcpyAsync(w.YD, w.XD, SD*4, hipMemcpyDeviceToDevice, stream);
    message(stream, mes1, w.rad1, w.rs1, edst, w);

    // ---- SubBlock 2 + energy head ----
    subblock(stream, sub2, w);
    gemm(stream, w.XA, lin_out, w.T1, BN, CA, CA, 1.f, 0);
    gemm(stream, w.T1, E1p,     w.T2, BN, CA, CA, 1.f, 0);
    k_head<<<g1(BN), 256, 0, stream>>>(w.T2, E2p, (float*)d_out);
}
